// InfoNCE_6425271075115
// MI455X (gfx1250) — matrix operand order repaired and checked by reading, not compiled, not run on hardware
//
#include <hip/hip_runtime.h>
#include <hip/hip_bf16.h>

// ---------------------------------------------------------------------------
// InfoNCE loss, fully fused.  n = 8192 (= 2 x 4096), d = 512.
//   loss = -(1/n) * sum_i [ dot_i / tau - L_i * labelsum_i ]
//   dot_i      = sim_pos + 0.75*(sum_top10 - (pos_in_top10 ? sim_pos : 0))
//   labelsum_i = 1 + 0.75*(pos_in_top10 ? 9 : 10)
//   L_i        = logsumexp_j( sim'[i,j]/tau ),  sim' has diag = -10
// Top-10 over cross-half columns only (values suffice; softmax is monotone,
// membership test is sim_pos >= 10th-largest value).
// Logits are bounded by 1/tau (unit vectors), so logsumexp uses the fixed
// max M0 = 1/tau: exact and branch-free (1 exp per element).
// ---------------------------------------------------------------------------

typedef __attribute__((ext_vector_type(16))) __bf16 v16bf;
typedef __attribute__((ext_vector_type(8)))  float  v8f;

#define N_TOT   8192
#define HALF    4096
#define DIM     512
#define INV_TAU (1.0f / 0.07f)
#define M0_LSE  INV_TAU
#define NEG_BIG (-1.0e30f)

union V16U { uint4 u4[2]; v16bf bf; };

__device__ __forceinline__ void insert10(float t[10], float v) {
  // keep t sorted descending; t holds the 10 largest values seen
  #pragma unroll
  for (int j = 0; j < 10; ++j) {
    float o   = t[j];
    bool  ins = v > o;
    t[j] = ins ? v : o;
    v    = ins ? o : v;
  }
}

// ---------------------------------------------------------------------------
// Kernel 1: row L2-normalize [z1;z2], split into bf16 hi/lo pair.
// 1024 blocks x 256 threads; one wave (32 lanes) per row.
// ---------------------------------------------------------------------------
__global__ void __launch_bounds__(256)
nrm_split(const float* __restrict__ z1, const float* __restrict__ z2,
          __bf16* __restrict__ hi, __bf16* __restrict__ lo) {
  int wave = threadIdx.x >> 5;
  int lane = threadIdx.x & 31;
  int row  = blockIdx.x * 8 + wave;
  const float* src = (row < HALF) ? (z1 + (size_t)row * DIM)
                                  : (z2 + (size_t)(row - HALF) * DIM);
  float vals[16];
  float ss = 0.f;
  #pragma unroll
  for (int j = 0; j < 16; ++j) {
    float v = src[lane + 32 * j];       // coalesced
    vals[j] = v;
    ss += v * v;
  }
  #pragma unroll
  for (int off = 16; off >= 1; off >>= 1) ss += __shfl_xor(ss, off);
  float inv = 1.0f / fmaxf(sqrtf(ss), 1e-12f);
  size_t base = (size_t)row * DIM;
  #pragma unroll
  for (int j = 0; j < 16; ++j) {
    float  x = vals[j] * inv;
    __bf16 h = (__bf16)x;
    __bf16 l = (__bf16)(x - (float)h);
    hi[base + lane + 32 * j] = h;
    lo[base + lane + 32 * j] = l;
  }
}

// ---------------------------------------------------------------------------
// Kernel 2: fused sim-GEMM + per-row stats.  One block per 16-row stripe.
// 8 waves; wave w handles column-tile PAIRS {2w, 2w+1} + 16*t, t = 0..31.
// A hi/lo operands are register-reused across the 2 column tiles:
// per K=32 chunk -> 12 b128 loads, 6 WMMAs (bf16x2 split: hi*hi+hi*lo+lo*hi).
// ---------------------------------------------------------------------------
__global__ void __launch_bounds__(256)
fused_sim(const __bf16* __restrict__ hi, const __bf16* __restrict__ lo,
          float* __restrict__ partial) {
  __shared__ float tile[8][2 * 16 * 17];   // per-wave two 16x16 tiles, pitch 17
  __shared__ float stats[256][12];         // per-lane {s, pos, top10[10]}

  int  wave   = threadIdx.x >> 5;
  int  lane   = threadIdx.x & 31;
  bool hiLane = lane >= 16;
  int  r      = lane & 15;
  int  row0   = blockIdx.x * 16;
  int  grow   = row0 + r;                  // global row this lane scans
  int  pcol   = grow ^ HALF;               // positive-pair column
  bool rowLow = row0 < HALF;

  const uint4* hi4 = (const uint4*)hi;     // 8 bf16 per uint4
  const uint4* lo4 = (const uint4*)lo;

  float s = 0.f, pos = NEG_BIG;
  float t10[10];
  #pragma unroll
  for (int j = 0; j < 10; ++j) t10[j] = NEG_BIG;

  // A-matrix 16x32 bf16 layout: lane<16 row M=lane holds K {kc..kc+7, kc+16..kc+23},
  // lane>=16 holds K {kc+8..kc+15, kc+24..kc+31}  (two 16-byte segments).
  size_t aBase = ((size_t)(row0 + r) * DIM) >> 3;   // uint4 units
  int    kAofs = hiLane ? 1 : 0;

  for (int tIdx = 0; tIdx < 32; ++tIdx) {
    int colbase0 = (2 * wave + 16 * tIdx) * 16;     // even tile of the pair
    // B-matrix 32x16: lane<16 col N=lane holds K {kc..kc+7, kc+16..kc+23},
    // lane>=16 holds K {kc+8..kc+15, kc+24..kc+31} (two 16-byte segments).
    size_t bBase0 = ((size_t)(colbase0 + r) * DIM) >> 3;
    size_t bBase1 = bBase0 + (16 * DIM >> 3);
    int    kBofs  = hiLane ? 1 : 0;

    v8f acc0 = {}, acc1 = {};
    for (int kc8 = 0; kc8 < DIM / 8; kc8 += 4) {    // K chunk of 32
      V16U aH, aL, bH0, bL0, bH1, bL1;
      size_t ai = aBase + kc8 + kAofs;
      aH.u4[0]  = hi4[ai];      aH.u4[1]  = hi4[ai + 2];
      aL.u4[0]  = lo4[ai];      aL.u4[1]  = lo4[ai + 2];
      size_t b0 = bBase0 + kc8 + kBofs;
      bH0.u4[0] = hi4[b0];      bH0.u4[1] = hi4[b0 + 2];
      bL0.u4[0] = lo4[b0];      bL0.u4[1] = lo4[b0 + 2];
      size_t b1 = bBase1 + kc8 + kBofs;
      bH1.u4[0] = hi4[b1];      bH1.u4[1] = hi4[b1 + 2];
      bL1.u4[0] = lo4[b1];      bL1.u4[1] = lo4[b1 + 2];
      acc0 = __builtin_amdgcn_wmma_f32_16x16x32_bf16(false, aH.bf, false, bH0.bf,
                                                     (short)0, acc0, false, false);
      acc1 = __builtin_amdgcn_wmma_f32_16x16x32_bf16(false, aH.bf, false, bH1.bf,
                                                     (short)0, acc1, false, false);
      acc0 = __builtin_amdgcn_wmma_f32_16x16x32_bf16(false, aH.bf, false, bL0.bf,
                                                     (short)0, acc0, false, false);
      acc1 = __builtin_amdgcn_wmma_f32_16x16x32_bf16(false, aH.bf, false, bL1.bf,
                                                     (short)0, acc1, false, false);
      acc0 = __builtin_amdgcn_wmma_f32_16x16x32_bf16(false, aL.bf, false, bH0.bf,
                                                     (short)0, acc0, false, false);
      acc1 = __builtin_amdgcn_wmma_f32_16x16x32_bf16(false, aL.bf, false, bH1.bf,
                                                     (short)0, acc1, false, false);
    }

    // Stage both tiles to LDS: lane holds col n=r, rows rr + (hiLane?8:0).
    float* tl = tile[wave];
    #pragma unroll
    for (int rr = 0; rr < 8; ++rr) {
      int mloc = rr + (hiLane ? 8 : 0);
      tl[mloc * 17 + r]       = acc0[rr];
      tl[272 + mloc * 17 + r] = acc1[rr];
    }
    __syncthreads();

    // Scan: lane pair (r, r+16) covers row r, 8 columns each, both tiles.
    int  c0        = hiLane ? 8 : 0;
    bool crossHalf = (rowLow != (colbase0 < HALF));  // pair never straddles halves
    #pragma unroll
    for (int half2 = 0; half2 < 2; ++half2) {
      const float* tb = tl + 272 * half2;
      int colbase = colbase0 + 16 * half2;
      #pragma unroll
      for (int c = 0; c < 8; ++c) {
        float v  = tb[r * 17 + c0 + c];
        int   gc = colbase + c0 + c;
        float vl = (gc == grow) ? -10.0f : v;        // diag fill
        s += __expf(vl * INV_TAU - M0_LSE);          // fixed-max logsumexp
        if (crossHalf && v > t10[9]) insert10(t10, v);
        pos = (gc == pcol) ? v : pos;
      }
    }
    __syncthreads();
  }

  // Dump per-lane partial stats.
  float* st = stats[wave * 32 + lane];
  st[0] = s; st[1] = pos;
  #pragma unroll
  for (int j = 0; j < 10; ++j) st[2 + j] = t10[j];
  __syncthreads();

  // Threads 0..15 merge the 16 partials of their row and compute loss_i.
  float loss = 0.f;
  if (threadIdx.x < 16) {
    int   rr = threadIdx.x;
    float S = 0.f, P = NEG_BIG;
    float T[10];
    #pragma unroll
    for (int j = 0; j < 10; ++j) T[j] = NEG_BIG;
    for (int w = 0; w < 8; ++w) {
      for (int h = 0; h < 2; ++h) {
        float* q = stats[w * 32 + h * 16 + rr];
        S += q[0];
        P  = fmaxf(P, q[1]);
        #pragma unroll
        for (int j = 0; j < 10; ++j) {
          float v = q[2 + j];
          if (v > T[9]) insert10(T, v);
        }
      }
    }
    float L     = M0_LSE + __logf(S);
    bool  in10  = (P >= T[9]);
    float sum10 = 0.f;
    #pragma unroll
    for (int j = 0; j < 10; ++j) sum10 += T[j];
    float dot    = P + 0.75f * (sum10 - (in10 ? P : 0.f));
    float lblsum = 1.0f + 0.75f * (in10 ? 9.0f : 10.0f);
    loss = -(dot * INV_TAU - L * lblsum);
  }
  #pragma unroll
  for (int off = 16; off >= 1; off >>= 1) loss += __shfl_xor(loss, off);
  if (threadIdx.x == 0) partial[blockIdx.x] = loss;
}

// ---------------------------------------------------------------------------
// Kernel 3: reduce 512 per-stripe partials -> scalar loss / n.
// ---------------------------------------------------------------------------
__global__ void __launch_bounds__(256)
final_reduce(const float* __restrict__ partial, float* __restrict__ out) {
  __shared__ float red[256];
  red[threadIdx.x] = partial[threadIdx.x] + partial[threadIdx.x + 256];
  __syncthreads();
  for (int off = 128; off >= 1; off >>= 1) {
    if (threadIdx.x < off) red[threadIdx.x] += red[threadIdx.x + off];
    __syncthreads();
  }
  if (threadIdx.x == 0) out[0] = red[0] * (1.0f / (float)N_TOT);
}

// ---------------------------------------------------------------------------
extern "C" void kernel_launch(void* const* d_in, const int* in_sizes, int n_in,
                              void* d_out, int out_size, void* d_ws, size_t ws_size,
                              hipStream_t stream) {
  const float* z1 = (const float*)d_in[0];
  const float* z2 = (const float*)d_in[1];

  __bf16* hi = (__bf16*)d_ws;                               //  8 MB
  __bf16* lo = hi + (size_t)N_TOT * DIM;                    //  8 MB
  float*  partial = (float*)((char*)d_ws +
                             2ull * N_TOT * DIM * sizeof(__bf16)); // 2 KB

  nrm_split   <<<N_TOT / 8,  256, 0, stream>>>(z1, z2, hi, lo);
  fused_sim   <<<N_TOT / 16, 256, 0, stream>>>(hi, lo, partial);
  final_reduce<<<1,          256, 0, stream>>>(partial, (float*)d_out);
}
